// GPSLayer_64484638982371
// MI455X (gfx1250) — matrix operand order repaired and checked by reading, not compiled, not run on hardware
//
#include <hip/hip_runtime.h>
#include <math.h>

#define N_NODES 50000
#define N_EDGES 500000
#define DIM     128
#define HEADS   4
#define LN_EPS  1e-5f

typedef __attribute__((ext_vector_type(16))) _Float16 v16h;
typedef __attribute__((ext_vector_type(8)))  _Float16 v8h;
typedef __attribute__((ext_vector_type(4)))  _Float16 v4h;
typedef __attribute__((ext_vector_type(8)))  float    v8f;

// ---------------- utility device functions ----------------

__device__ __forceinline__ float waveSum(float v) {
#pragma unroll
  for (int i = 16; i > 0; i >>= 1) v += __shfl_xor(v, i, 32);
  return v;
}

// monotonic float atomic max via int trick (init must be -inf)
__device__ __forceinline__ void atomicMaxF(float* addr, float val) {
  if (val >= 0.0f) {
    atomicMax((int*)addr, __float_as_int(val));
  } else {
    atomicMin((unsigned int*)addr, __float_as_uint(val));
  }
}

// ---------------- small elementwise kernels ----------------

__global__ void fill_f32(float* __restrict__ p, float v, int n) {
  int t = blockIdx.x * blockDim.x + threadIdx.x;
  if (t < n) p[t] = v;
}

__global__ void cvt_f32_f16(const float* __restrict__ in, _Float16* __restrict__ out, int n) {
  int t = blockIdx.x * blockDim.x + threadIdx.x;
  if (t < n) out[t] = (_Float16)in[t];
}

__global__ void relu_cvt_f16(const float* __restrict__ in, _Float16* __restrict__ out, int n) {
  int t = blockIdx.x * blockDim.x + threadIdx.x;
  if (t < n) out[t] = (_Float16)fmaxf(in[t], 0.0f);
}

// transpose + convert weight: W[dIn,dOut] f32 -> Wt[dOut,dIn] f16
__global__ void wprep(const float* __restrict__ W, _Float16* __restrict__ Wt, int dIn, int dOut) {
  int t = blockIdx.x * blockDim.x + threadIdx.x;
  if (t >= dIn * dOut) return;
  int kI = t / dOut, n = t % dOut;
  Wt[(size_t)n * dIn + kI] = (_Float16)W[t];
}

// ---------------- WMMA GEMM:  C[N,dOut] = A[N,dIn](f16) @ Wt^T + bias -------
// one wave computes one 16x16 output tile; 8 waves (256 thr) per block
// grid.x = row tiles (N/16), grid.y = ceil(colTiles/8)

__global__ __launch_bounds__(256) void gemm_wmma(
    const _Float16* __restrict__ A, const _Float16* __restrict__ Wt,
    const float* __restrict__ bias, float* __restrict__ C,
    int dIn, int dOut) {
  int l = threadIdx.x & 31;
  int wave = threadIdx.x >> 5;
  int colTile = blockIdx.y * 8 + wave;
  if (colTile * 16 >= dOut) return;           // wave-uniform exit, EXEC stays full
  int rowTile = blockIdx.x;

  int m   = l & 15;
  int k0a = (l >> 4) << 3;                    // A: half-rows at k0a, k0a+16
  int k0b = (l >> 4) << 3;                    // B: half-cols at k0b, k0b+16
  const _Float16* arow = A  + (size_t)(rowTile * 16 + m) * dIn;
  const _Float16* bcol = Wt + (size_t)(colTile * 16 + (l & 15)) * dIn + k0b;

  union { v16h v; v8h h[2]; } af, bf;
  v8f acc = {};
  for (int kk = 0; kk < dIn; kk += 32) {
    af.h[0] = *(const v8h*)(arow + kk + k0a);
    af.h[1] = *(const v8h*)(arow + kk + k0a + 16);
    bf.h[0] = *(const v8h*)(bcol + kk);
    bf.h[1] = *(const v8h*)(bcol + kk + 16);
    acc = __builtin_amdgcn_wmma_f32_16x16x32_f16(
        /*neg_a=*/false, af.v, /*neg_b=*/false, bf.v,
        /*c_mod=*/(short)0, acc, /*reuse_a=*/false, /*reuse_b=*/false);
  }

  int n = colTile * 16 + (l & 15);
  float bv = bias ? bias[n] : 0.0f;
  int mbase = rowTile * 16 + ((l >> 4) << 3);
#pragma unroll
  for (int r = 0; r < 8; ++r)
    C[(size_t)(mbase + r) * dOut + n] = acc[r] + bv;
}

// ---------------- graph kernels ----------------

__global__ void deg_kernel(const int* __restrict__ dst, float* __restrict__ deg, int E) {
  int e = blockIdx.x * blockDim.x + threadIdx.x;
  if (e < E) atomicAdd(deg + dst[e], 1.0f);
}

__global__ void dinv_kernel(float* __restrict__ d, int n) {
  int t = blockIdx.x * blockDim.x + threadIdx.x;
  if (t < n) { float v = d[t]; d[t] = (v > 0.0f) ? rsqrtf(fmaxf(v, 1.0f)) : 0.0f; }
}

// GCN scatter: thread = (edge, 4-float chunk); 32 chunks cover 128 channels
__global__ void gcn_scatter(const int* __restrict__ src, const int* __restrict__ dst,
                            const float* __restrict__ xw, const float* __restrict__ dinv,
                            float* __restrict__ acc, int E) {
  int t = blockIdx.x * blockDim.x + threadIdx.x;
  if (t >= E * 32) return;
  int e = t >> 5, ch = (t & 31) << 2;
  int s = src[e], d = dst[e];
  float c = dinv[s] * dinv[d];
  float4 xv = *(const float4*)(xw + (size_t)s * DIM + ch);
  float* ap = acc + (size_t)d * DIM + ch;
  atomicAdd(ap + 0, c * xv.x); atomicAdd(ap + 1, c * xv.y);
  atomicAdd(ap + 2, c * xv.z); atomicAdd(ap + 3, c * xv.w);
}

// attention logits: one wave per edge; 4 heads of C=32; lanes l cover chans l*4..l*4+3
__global__ void alpha_kernel(const int* __restrict__ src, const int* __restrict__ dst,
                             const float* __restrict__ q, const float* __restrict__ k,
                             float* __restrict__ alphaB, float* __restrict__ mB, int E) {
  int wid = blockIdx.x * (blockDim.x >> 5) + (threadIdx.x >> 5);
  if (wid >= E) return;
  int l = threadIdx.x & 31;
  int s = src[wid], d = dst[wid];
  float4 qv = *(const float4*)(q + (size_t)d * DIM + l * 4);
  float4 kv = *(const float4*)(k + (size_t)s * DIM + l * 4);
  float p = qv.x * kv.x + qv.y * kv.y + qv.z * kv.z + qv.w * kv.w;
  p += __shfl_xor(p, 1, 32);
  p += __shfl_xor(p, 2, 32);
  p += __shfl_xor(p, 4, 32);                 // reduce within each 8-lane head group
  if ((l & 7) == 0) {
    int hd = l >> 3;
    float al = p * 0.17677669529663689f;     // 1/sqrt(32)
    alphaB[(size_t)wid * HEADS + hd] = al;
    atomicMaxF(mB + (size_t)d * HEADS + hd, al);
  }
}

__global__ void mfix_kernel(float* __restrict__ m, int n) {
  int t = blockIdx.x * blockDim.x + threadIdx.x;
  if (t < n) { float v = m[t]; m[t] = isinf(v) ? 0.0f : v; }
}

__global__ void exp_kernel(const int* __restrict__ dst, float* __restrict__ alphaB,
                           const float* __restrict__ mB, float* __restrict__ sB, int n) {
  int t = blockIdx.x * blockDim.x + threadIdx.x;
  if (t >= n) return;
  int e = t >> 2, hd = t & 3;
  int d = dst[e];
  float a = expf(alphaB[t] - mB[(size_t)d * HEADS + hd]);
  alphaB[t] = a;
  atomicAdd(sB + (size_t)d * HEADS + hd, a);
}

__global__ void agg_scatter(const int* __restrict__ src, const int* __restrict__ dst,
                            const float* __restrict__ v, const float* __restrict__ alphaB,
                            const float* __restrict__ sB, float* __restrict__ agg, int E) {
  int t = blockIdx.x * blockDim.x + threadIdx.x;
  if (t >= E * 32) return;
  int e = t >> 5, c4 = (t & 31);
  int hd = c4 >> 3;
  int s = src[e], d = dst[e];
  float w = alphaB[(size_t)e * HEADS + hd] /
            fmaxf(sB[(size_t)d * HEADS + hd], 1e-16f);
  int ch = c4 << 2;
  float4 vv = *(const float4*)(v + (size_t)s * DIM + ch);
  float* ap = agg + (size_t)d * DIM + ch;
  atomicAdd(ap + 0, w * vv.x); atomicAdd(ap + 1, w * vv.y);
  atomicAdd(ap + 2, w * vv.z); atomicAdd(ap + 3, w * vv.w);
}

// ---------------- per-node epilogues (one wave per node) ----------------

__global__ void combine_ln1(const float* __restrict__ localAcc, const float* __restrict__ bgcn,
                            const float* __restrict__ agg, const float* __restrict__ xr,
                            const float* __restrict__ wbeta,
                            const float* __restrict__ g1, const float* __restrict__ b1,
                            const float* __restrict__ localW, const float* __restrict__ globalW,
                            float* __restrict__ hbuf, _Float16* __restrict__ hh, int N) {
  int wid = blockIdx.x * (blockDim.x >> 5) + (threadIdx.x >> 5);
  if (wid >= N) return;
  int l = threadIdx.x & 31;
  size_t base = (size_t)wid * DIM + l * 4;
  float4 og = *(const float4*)(agg + base);
  float4 xv = *(const float4*)(xr + base);
  float4 lo = *(const float4*)(localAcc + base);
  float4 bg = *(const float4*)(bgcn + l * 4);
  lo.x += bg.x; lo.y += bg.y; lo.z += bg.z; lo.w += bg.w;

  float4 w1 = *(const float4*)(wbeta + l * 4);
  float4 w2 = *(const float4*)(wbeta + 128 + l * 4);
  float4 w3 = *(const float4*)(wbeta + 256 + l * 4);
  float p = og.x * w1.x + og.y * w1.y + og.z * w1.z + og.w * w1.w
          + xv.x * w2.x + xv.y * w2.y + xv.z * w2.z + xv.w * w2.w
          + (og.x - xv.x) * w3.x + (og.y - xv.y) * w3.y
          + (og.z - xv.z) * w3.z + (og.w - xv.w) * w3.w;
  p = waveSum(p);
  float beta = 1.0f / (1.0f + expf(-p));
  float lw = localW[0], gw = globalW[0];

  float4 h;
  h.x = lw * lo.x + gw * (beta * xv.x + (1.0f - beta) * og.x);
  h.y = lw * lo.y + gw * (beta * xv.y + (1.0f - beta) * og.y);
  h.z = lw * lo.z + gw * (beta * xv.z + (1.0f - beta) * og.z);
  h.w = lw * lo.w + gw * (beta * xv.w + (1.0f - beta) * og.w);

  float4 y; y.x = h.x + h.x; y.y = h.y + h.y; y.z = h.z + h.z; y.w = h.w + h.w;
  float mu = waveSum(y.x + y.y + y.z + y.w) * (1.0f / DIM);
  float d0 = y.x - mu, d1 = y.y - mu, d2 = y.z - mu, d3 = y.w - mu;
  float var = waveSum(d0 * d0 + d1 * d1 + d2 * d2 + d3 * d3) * (1.0f / DIM);
  float inv = rsqrtf(var + LN_EPS);
  float4 g = *(const float4*)(g1 + l * 4);
  float4 b = *(const float4*)(b1 + l * 4);
  float4 o;
  o.x = d0 * inv * g.x + b.x; o.y = d1 * inv * g.y + b.y;
  o.z = d2 * inv * g.z + b.z; o.w = d3 * inv * g.w + b.w;
  *(float4*)(hbuf + base) = o;
  v4h oh = { (_Float16)o.x, (_Float16)o.y, (_Float16)o.z, (_Float16)o.w };
  *(v4h*)(hh + base) = oh;
}

__global__ void add_ln2(const float* __restrict__ ffn2, const float* __restrict__ hbuf,
                        const float* __restrict__ g2, const float* __restrict__ b2,
                        float* __restrict__ out, int N) {
  int wid = blockIdx.x * (blockDim.x >> 5) + (threadIdx.x >> 5);
  if (wid >= N) return;
  int l = threadIdx.x & 31;
  size_t base = (size_t)wid * DIM + l * 4;
  float4 a = *(const float4*)(ffn2 + base);
  float4 h = *(const float4*)(hbuf + base);
  float4 y; y.x = a.x + h.x; y.y = a.y + h.y; y.z = a.z + h.z; y.w = a.w + h.w;
  float mu = waveSum(y.x + y.y + y.z + y.w) * (1.0f / DIM);
  float d0 = y.x - mu, d1 = y.y - mu, d2 = y.z - mu, d3 = y.w - mu;
  float var = waveSum(d0 * d0 + d1 * d1 + d2 * d2 + d3 * d3) * (1.0f / DIM);
  float inv = rsqrtf(var + LN_EPS);
  float4 g = *(const float4*)(g2 + l * 4);
  float4 b = *(const float4*)(b2 + l * 4);
  float4 o;
  o.x = d0 * inv * g.x + b.x; o.y = d1 * inv * g.y + b.y;
  o.z = d2 * inv * g.z + b.z; o.w = d3 * inv * g.w + b.w;
  *(float4*)(out + base) = o;
}

// ---------------- host launch ----------------

static inline size_t alignUp(size_t x) { return (x + 255) & ~size_t(255); }
static inline int cdiv(long a, long b) { return (int)((a + b - 1) / b); }

extern "C" void kernel_launch(void* const* d_in, const int* in_sizes, int n_in,
                              void* d_out, int out_size, void* d_ws, size_t ws_size,
                              hipStream_t stream) {
  const float* x       = (const float*)d_in[0];
  const int*   ei      = (const int*)d_in[1];
  const int*   src     = ei;
  const int*   dst     = ei + N_EDGES;
  const float* w_gcn   = (const float*)d_in[2];
  const float* b_gcn   = (const float*)d_in[3];
  const float* wq      = (const float*)d_in[4];
  const float* bq      = (const float*)d_in[5];
  const float* wk      = (const float*)d_in[6];
  const float* bk      = (const float*)d_in[7];
  const float* wv      = (const float*)d_in[8];
  const float* bv      = (const float*)d_in[9];
  const float* w_skip  = (const float*)d_in[10];
  const float* b_skip  = (const float*)d_in[11];
  const float* w_beta  = (const float*)d_in[12];
  const float* g1      = (const float*)d_in[13];
  const float* b1      = (const float*)d_in[14];
  const float* g2      = (const float*)d_in[15];
  const float* b2      = (const float*)d_in[16];
  const float* w_rel   = (const float*)d_in[17];
  const float* w_root  = (const float*)d_in[18];
  const float* localW  = (const float*)d_in[19];
  const float* globalW = (const float*)d_in[20];
  float* out = (float*)d_out;

  // ---- workspace layout ----
  char* p = (char*)d_ws;
  size_t off = 0;
  auto take = [&](size_t bytes) { char* r = p + off; off = alignUp(off + bytes); return r; };

  _Float16* xh    = (_Float16*)take((size_t)N_NODES * DIM * 2);
  _Float16* wTg   = (_Float16*)take(128 * 128 * 2);
  _Float16* wTq   = (_Float16*)take(128 * 128 * 2);
  _Float16* wTk   = (_Float16*)take(128 * 128 * 2);
  _Float16* wTv   = (_Float16*)take(128 * 128 * 2);
  _Float16* wTs   = (_Float16*)take(128 * 128 * 2);
  _Float16* wTrel = (_Float16*)take(256 * 128 * 2);
  _Float16* wTroot= (_Float16*)take(256 * 128 * 2);
  float* XW   = (float*)take((size_t)N_NODES * DIM * 4);   // later reused as FFN2
  float* Q    = (float*)take((size_t)N_NODES * DIM * 4);   // Q+K later reused as FFN1
  float* K    = (float*)take((size_t)N_NODES * DIM * 4);
  float* V    = (float*)take((size_t)N_NODES * DIM * 4);   // later reused as FFN1H (f16)
  float* XR   = (float*)take((size_t)N_NODES * DIM * 4);
  float* LACC = (float*)take((size_t)N_NODES * DIM * 4);
  float* AGG  = (float*)take((size_t)N_NODES * DIM * 4);
  float* HBUF = (float*)take((size_t)N_NODES * DIM * 4);
  _Float16* HH = (_Float16*)take((size_t)N_NODES * DIM * 2);
  float* DINV = (float*)take((size_t)N_NODES * 4);
  float* ALPH = (float*)take((size_t)N_EDGES * HEADS * 4);
  float* MB   = (float*)take((size_t)N_NODES * HEADS * 4);
  float* SB   = (float*)take((size_t)N_NODES * HEADS * 4);
  float*    FFN1  = Q;                    // [N,256] f32 spans Q..K
  _Float16* FFN1H = (_Float16*)V;         // [N,256] f16 fits in V
  float*    FFN2  = XW;                   // [N,128] f32

  const int T = 256;
  const int rowTiles = N_NODES / 16;      // 3125 exactly

  // 1. convert x to f16
  cvt_f32_f16<<<cdiv((long)N_NODES * DIM, T), T, 0, stream>>>(x, xh, N_NODES * DIM);

  // 2. prep transposed f16 weights
  wprep<<<cdiv(128 * 128, T), T, 0, stream>>>(w_gcn,  wTg,   128, 128);
  wprep<<<cdiv(128 * 128, T), T, 0, stream>>>(wq,     wTq,   128, 128);
  wprep<<<cdiv(128 * 128, T), T, 0, stream>>>(wk,     wTk,   128, 128);
  wprep<<<cdiv(128 * 128, T), T, 0, stream>>>(wv,     wTv,   128, 128);
  wprep<<<cdiv(128 * 128, T), T, 0, stream>>>(w_skip, wTs,   128, 128);
  wprep<<<cdiv(128 * 256, T), T, 0, stream>>>(w_rel,  wTrel, 128, 256);
  wprep<<<cdiv(256 * 128, T), T, 0, stream>>>(w_root, wTroot,256, 128);

  // 3. dense GEMMs (WMMA f16->f32)
  dim3 g128(rowTiles, 1), g256(rowTiles, 2);
  gemm_wmma<<<g128, T, 0, stream>>>(xh, wTg, nullptr, XW, 128, 128);
  gemm_wmma<<<g128, T, 0, stream>>>(xh, wTq, bq,      Q,  128, 128);
  gemm_wmma<<<g128, T, 0, stream>>>(xh, wTk, bk,      K,  128, 128);
  gemm_wmma<<<g128, T, 0, stream>>>(xh, wTv, bv,      V,  128, 128);
  gemm_wmma<<<g128, T, 0, stream>>>(xh, wTs, b_skip,  XR, 128, 128);

  // 4. degree + symmetric norm
  fill_f32<<<cdiv(N_NODES, T), T, 0, stream>>>(DINV, 0.0f, N_NODES);
  deg_kernel<<<cdiv(N_EDGES, T), T, 0, stream>>>(dst, DINV, N_EDGES);
  dinv_kernel<<<cdiv(N_NODES, T), T, 0, stream>>>(DINV, N_NODES);

  // 5. GCN aggregation
  fill_f32<<<cdiv((long)N_NODES * DIM, T), T, 0, stream>>>(LACC, 0.0f, N_NODES * DIM);
  gcn_scatter<<<cdiv((long)N_EDGES * 32, T), T, 0, stream>>>(src, dst, XW, DINV, LACC, N_EDGES);

  // 6. segment softmax attention
  fill_f32<<<cdiv(N_NODES * HEADS, T), T, 0, stream>>>(MB, -INFINITY, N_NODES * HEADS);
  alpha_kernel<<<cdiv(N_EDGES, 8), T, 0, stream>>>(src, dst, Q, K, ALPH, MB, N_EDGES);
  mfix_kernel<<<cdiv(N_NODES * HEADS, T), T, 0, stream>>>(MB, N_NODES * HEADS);
  fill_f32<<<cdiv(N_NODES * HEADS, T), T, 0, stream>>>(SB, 0.0f, N_NODES * HEADS);
  exp_kernel<<<cdiv((long)N_EDGES * HEADS, T), T, 0, stream>>>(dst, ALPH, MB, SB, N_EDGES * HEADS);
  fill_f32<<<cdiv((long)N_NODES * DIM, T), T, 0, stream>>>(AGG, 0.0f, N_NODES * DIM);
  agg_scatter<<<cdiv((long)N_EDGES * 32, T), T, 0, stream>>>(src, dst, V, ALPH, SB, AGG, N_EDGES);

  // 7. beta gate + combine + LayerNorm1
  combine_ln1<<<cdiv(N_NODES, 8), T, 0, stream>>>(LACC, b_gcn, AGG, XR, w_beta,
                                                  g1, b1, localW, globalW, HBUF, HH, N_NODES);

  // 8. FFN: relu(h@w_rel)@w_root, then residual LayerNorm2 -> d_out
  gemm_wmma<<<g256, T, 0, stream>>>(HH, wTrel, nullptr, FFN1, 128, 256);
  relu_cvt_f16<<<cdiv((long)N_NODES * 256, T), T, 0, stream>>>(FFN1, FFN1H, N_NODES * 256);
  gemm_wmma<<<g128, T, 0, stream>>>(FFN1H, wTroot, nullptr, FFN2, 256, 128);
  add_ln2<<<cdiv(N_NODES, 8), T, 0, stream>>>(FFN2, HBUF, g2, b2, out, N_NODES);

  (void)in_sizes; (void)n_in; (void)out_size; (void)ws_size;
}
